// get_model_70643622084666
// MI455X (gfx1250) — matrix operand order repaired and checked by reading, not compiled, not run on hardware
//
#include <hip/hip_runtime.h>

// PointNet++ (SSG) forward for MI455X (gfx1250, wave32).
// Heavy shared-MLP layers run on v_wmma_f32_16x16x32_f16 (f16 in, f32 acc),
// register-blocked 1x4 N-tiles per wave for 4x A-fragment reuse.
// Irregular stages (FPS/ball-query/grouping/BN stats/head) are plain kernels.

typedef _Float16 half_t;
typedef __attribute__((ext_vector_type(8)))  _Float16 v8h;
typedef __attribute__((ext_vector_type(16))) _Float16 v16h;
typedef __attribute__((ext_vector_type(8)))  float    v8f;

#define BN_EPS 1e-5f

// ---------------- utility kernels ----------------

__global__ void cvt_f32_f16_kernel(const float* __restrict__ x, half_t* __restrict__ y, int n) {
  int t = blockIdx.x * blockDim.x + threadIdx.x;
  if (t < n) y[t] = (half_t)x[t];
}

// xyz (B,3,N) -> pts (B,N,3)
__global__ void transpose_kernel(const float* __restrict__ xyz, float* __restrict__ pts,
                                 int Bb, int N) {
  int t = blockIdx.x * blockDim.x + threadIdx.x;
  if (t >= Bb * N) return;
  int b = t / N, n = t % N;
  const float* src = xyz + (size_t)b * 3 * N;
  float* dst = pts + (size_t)t * 3;
  dst[0] = src[n];
  dst[1] = src[N + n];
  dst[2] = src[2 * N + n];
}

// Farthest point sampling: one block (512 threads) per batch.
// Matches reference scan: out[0]=0, dist=min(dist,d(centroid)), next=argmax(first max).
template <int NPTS>
__global__ void fps_kernel(const float* __restrict__ xyz, int* __restrict__ idx_out, int npoint) {
  constexpr int PER = (NPTS + 511) / 512;
  int b = blockIdx.x;
  const float* p = xyz + (size_t)b * NPTS * 3;
  __shared__ float sval[512];
  __shared__ int   sidx[512];
  __shared__ float cent[3];
  float dist[PER];
#pragma unroll
  for (int j = 0; j < PER; ++j) dist[j] = 1e10f;
  int far = 0;
  for (int it = 0; it < npoint; ++it) {
    if (threadIdx.x == 0) idx_out[b * npoint + it] = far;
    if (threadIdx.x < 3) cent[threadIdx.x] = p[far * 3 + threadIdx.x];
    __syncthreads();
    float cx = cent[0], cy = cent[1], cz = cent[2];
    float best = -1.0f;
    int bi = NPTS;
#pragma unroll
    for (int j = 0; j < PER; ++j) {
      int n = threadIdx.x + j * 512;
      if (n < NPTS) {
        float dx = p[n * 3 + 0] - cx, dy = p[n * 3 + 1] - cy, dz = p[n * 3 + 2] - cz;
        float d = dx * dx + dy * dy + dz * dz;
        d = fminf(dist[j], d);
        dist[j] = d;
        if (d > best) { best = d; bi = n; }   // strict > keeps earliest index
      }
    }
    sval[threadIdx.x] = best;
    sidx[threadIdx.x] = bi;
    __syncthreads();
    for (int s = 256; s > 0; s >>= 1) {
      if (threadIdx.x < s) {
        float v1 = sval[threadIdx.x], v2 = sval[threadIdx.x + s];
        int   i1 = sidx[threadIdx.x], i2 = sidx[threadIdx.x + s];
        if (v2 > v1 || (v2 == v1 && i2 < i1)) { sval[threadIdx.x] = v2; sidx[threadIdx.x] = i2; }
      }
      __syncthreads();
    }
    far = sidx[0];
    __syncthreads();
  }
}

// out[b,s,:] = xyz[b, fidx[b,s], :]
__global__ void gather_xyz_kernel(const float* __restrict__ xyz, const int* __restrict__ fidx,
                                  float* __restrict__ out, int Npts, int S, int total) {
  int t = blockIdx.x * blockDim.x + threadIdx.x;
  if (t >= total) return;
  int b = t / S;
  int j = fidx[t];
  const float* src = xyz + ((size_t)b * Npts + j) * 3;
  out[(size_t)t * 3 + 0] = src[0];
  out[(size_t)t * 3 + 1] = src[1];
  out[(size_t)t * 3 + 2] = src[2];
}

// Ball query: thread per (b,s); in-order scan == sorted-ascending-index semantics.
__global__ void ballquery_kernel(const float* __restrict__ xyz, const float* __restrict__ nxyz,
                                 int* __restrict__ idx_out, int Bb, int Npts, int S,
                                 int nsample, float r2) {
  int t = blockIdx.x * blockDim.x + threadIdx.x;
  if (t >= Bb * S) return;
  int b = t / S;
  const float* p = xyz + (size_t)b * Npts * 3;
  float qx = nxyz[(size_t)t * 3 + 0], qy = nxyz[(size_t)t * 3 + 1], qz = nxyz[(size_t)t * 3 + 2];
  int* out = idx_out + (size_t)t * nsample;
  int cnt = 0;
  for (int n = 0; n < Npts && cnt < nsample; ++n) {
    float dx = p[n * 3 + 0] - qx, dy = p[n * 3 + 1] - qy, dz = p[n * 3 + 2] - qz;
    if (dx * dx + dy * dy + dz * dz <= r2) out[cnt++] = n;
  }
  int first = (cnt > 0) ? out[0] : 0;
  for (int k = cnt; k < nsample; ++k) out[k] = first;
}

// sa1 grouping: centered xyz only (3 channels), f16 output
__global__ void group1_kernel(const float* __restrict__ pts, const float* __restrict__ nxyz,
                              const int* __restrict__ idx, half_t* __restrict__ Gout,
                              int N, int S, int K, int total) {
  int t = blockIdx.x * blockDim.x + threadIdx.x;
  if (t >= total) return;
  int b = t / (S * K);
  int s = (t / K) % S;
  int j = idx[t];
  const float* p = pts + ((size_t)b * N + j) * 3;
  const float* q = nxyz + ((size_t)b * S + s) * 3;
  half_t* g = Gout + (size_t)t * 3;
  g[0] = (half_t)(p[0] - q[0]);
  g[1] = (half_t)(p[1] - q[1]);
  g[2] = (half_t)(p[2] - q[2]);
}

// sa2 grouping: [centered xyz (3) | gathered features (Cpts)] -> f16, stride 3+Cpts
__global__ void group2_kernel(const float* __restrict__ l1xyz, const float* __restrict__ l2xyz,
                              const float* __restrict__ l1pts, const int* __restrict__ idx,
                              half_t* __restrict__ Gout, int N, int S, int K, int Cpts,
                              int total) {
  int t = blockIdx.x * blockDim.x + threadIdx.x;
  if (t >= total) return;
  int b = t / (S * K);
  int s = (t / K) % S;
  int j = idx[t];
  const float* p = l1xyz + ((size_t)b * N + j) * 3;
  const float* q = l2xyz + ((size_t)b * S + s) * 3;
  half_t* g = Gout + (size_t)t * (3 + Cpts);
  g[0] = (half_t)(p[0] - q[0]);
  g[1] = (half_t)(p[1] - q[1]);
  g[2] = (half_t)(p[2] - q[2]);
  const float* f = l1pts + ((size_t)b * N + j) * Cpts;
  for (int c = 0; c < Cpts; ++c) g[3 + c] = (half_t)f[c];
}

// sa3 group_all: [raw xyz (3) | features (Cpts)] elementwise, no centering
__global__ void group3_kernel(const float* __restrict__ l2xyz, const float* __restrict__ l2pts,
                              half_t* __restrict__ Gout, int BK, int Cpts) {
  int C = 3 + Cpts;
  int t = blockIdx.x * blockDim.x + threadIdx.x;
  if (t >= BK * C) return;
  int row = t / C, c = t % C;
  float v = (c < 3) ? l2xyz[(size_t)row * 3 + c] : l2pts[(size_t)row * Cpts + (c - 3)];
  Gout[t] = (half_t)v;
}

// Plain GEMM (row-major X[R,C] * W[O,C]^T + bias), templated on in/out element types.
template <typename TI, typename TO>
__global__ void lin_kernel(const TI* __restrict__ X, const float* __restrict__ W,
                           const float* __restrict__ bias, TO* __restrict__ Y,
                           int R, int C, int O) {
  int t = blockIdx.x * blockDim.x + threadIdx.x;
  if (t >= R * O) return;
  int r = t / O, o = t % O;
  const TI* x = X + (size_t)r * C;
  const float* w = W + (size_t)o * C;
  float s = bias[o];
  for (int c = 0; c < C; ++c) s += (float)x[c] * w[c];
  Y[t] = (TO)s;
}

// WMMA GEMM: Y[R,O] = X[R,C] * Wh[O,C]^T + bias.
// R mult of 16; O mult of 64; C mult of 32.
// One 16x64 super-tile per wave (4 accumulators, A-fragment reused 4x);
// 8 waves (256 threads) per block.
__global__ void gemm_wmma_kernel(const half_t* __restrict__ X, const half_t* __restrict__ Wh,
                                 const float* __restrict__ bias, half_t* __restrict__ Y,
                                 int R, int C, int O) {
  int wave = threadIdx.x >> 5;
  int lane = threadIdx.x & 31;
  int mtiles = R >> 4;
  int n4tiles = O >> 6;                       // super-tiles of 64 output cols
  long st = (long)blockIdx.x * 8 + wave;
  if (st >= (long)mtiles * n4tiles) return;   // wave-uniform: EXEC stays all-ones
  int tm  = (int)(st % mtiles);
  int tn4 = (int)(st / mtiles);
  int lh = lane & 15;                         // row (A) / col (B,D) within tile
  int hh = lane >> 4;                         // K-half selector
  // A layout: lane holds row lh; halves K = 8*hh.. +7 and K = 8*hh+16.. +23
  const half_t* xrow = X + (size_t)(tm * 16 + lh) * C + hh * 8;
  // B layout: lane holds column lh; halves K = 8*hh.. +7 and K = 8*hh+16.. +23
  const half_t* wbase = Wh + (size_t)(tn4 * 64 + lh) * C + hh * 8;
  const size_t wstride = (size_t)16 * C;      // next 16-column tile of W
  v8f acc[4] = {{}, {}, {}, {}};
  for (int k0 = 0; k0 < C; k0 += 32) {
    v8h a0 = *(const v8h*)(xrow + k0);
    v8h a1 = *(const v8h*)(xrow + k0 + 16);
    // speculative prefetch of the next K-step of the A stream (global_prefetch_b8)
    __builtin_prefetch(xrow + k0 + 32, 0, 0);
    v16h a;
#pragma unroll
    for (int i = 0; i < 8; ++i) { a[i] = a0[i]; a[i + 8] = a1[i]; }
#pragma unroll
    for (int j = 0; j < 4; ++j) {
      const half_t* w = wbase + (size_t)j * wstride + k0;
      v8h b0 = *(const v8h*)(w);
      v8h b1 = *(const v8h*)(w + 16);
      v16h bm;
#pragma unroll
      for (int i = 0; i < 8; ++i) { bm[i] = b0[i]; bm[i + 8] = b1[i]; }
      acc[j] = __builtin_amdgcn_wmma_f32_16x16x32_f16(false, a, false, bm,
                                                      (short)0, acc[j], false, false);
    }
  }
  // D layout: VGPR i -> row i + 8*hh, column lh
  int rbase = tm * 16 + hh * 8;
#pragma unroll
  for (int j = 0; j < 4; ++j) {
    int ocol = tn4 * 64 + j * 16 + lh;
    float bo = bias[ocol];
#pragma unroll
    for (int i = 0; i < 8; ++i)
      Y[(size_t)(rbase + i) * O + ocol] = (half_t)(acc[j][i] + bo);
  }
}

// Per-channel mean/var (biased) over R rows; one block per channel.
__global__ void stats_kernel(const half_t* __restrict__ X, float* __restrict__ mean,
                             float* __restrict__ var, int R, int O) {
  int o = blockIdx.x;
  __shared__ float ssum[256];
  __shared__ float ssq[256];
  float s = 0.f, q = 0.f;
  for (int r = threadIdx.x; r < R; r += 256) {
    float x = (float)X[(size_t)r * O + o];
    s += x;
    q += x * x;
  }
  ssum[threadIdx.x] = s;
  ssq[threadIdx.x] = q;
  __syncthreads();
  for (int st = 128; st > 0; st >>= 1) {
    if (threadIdx.x < st) {
      ssum[threadIdx.x] += ssum[threadIdx.x + st];
      ssq[threadIdx.x]  += ssq[threadIdx.x + st];
    }
    __syncthreads();
  }
  if (threadIdx.x == 0) {
    float m = ssum[0] / (float)R;
    mean[o] = m;
    var[o] = ssq[0] / (float)R - m * m;
  }
}

// In-place BN + ReLU on f16 activations
__global__ void bnrelu_kernel(half_t* __restrict__ X, const float* __restrict__ mean,
                              const float* __restrict__ var, const float* __restrict__ g,
                              const float* __restrict__ beta, long total, int O) {
  long t = (long)blockIdx.x * blockDim.x + threadIdx.x;
  if (t >= total) return;
  int o = (int)(t % O);
  float x = (float)X[t];
  float y = (x - mean[o]) * rsqrtf(var[o] + BN_EPS) * g[o] + beta[o];
  X[t] = (half_t)fmaxf(y, 0.f);
}

// Max-pool over sample dim: X (BS, K, O) f16 -> Y (BS, O) f32
__global__ void pool_kernel(const half_t* __restrict__ X, float* __restrict__ Y,
                            int BS, int K, int O) {
  int t = blockIdx.x * blockDim.x + threadIdx.x;
  if (t >= BS * O) return;
  int bs = t / O, o = t % O;
  const half_t* x = X + (size_t)bs * K * O + o;
  float m = -3.4e38f;
  for (int k = 0; k < K; ++k) m = fmaxf(m, (float)x[(size_t)k * O]);
  Y[t] = m;
}

__global__ void logsoftmax_kernel(const float* __restrict__ X, float* __restrict__ Y,
                                  int R, int C) {
  int r = blockIdx.x * blockDim.x + threadIdx.x;
  if (r >= R) return;
  const float* x = X + (size_t)r * C;
  float mx = x[0];
  for (int c = 1; c < C; ++c) mx = fmaxf(mx, x[c]);
  float s = 0.f;
  for (int c = 0; c < C; ++c) s += expf(x[c] - mx);
  float lse = mx + logf(s);
  for (int c = 0; c < C; ++c) Y[(size_t)r * C + c] = x[c] - lse;
}

// ---------------- host orchestration ----------------

extern "C" void kernel_launch(void* const* d_in, const int* in_sizes, int n_in,
                              void* d_out, int out_size, void* d_ws, size_t ws_size,
                              hipStream_t stream) {
  (void)in_sizes; (void)n_in; (void)out_size; (void)ws_size;
  const int Bb = 32, N = 2048;
  const int S1 = 512, K1 = 32;
  const int S2 = 128, K2 = 64;
  const int K3 = 128;
  const int R1 = Bb * S1 * K1;   // 524288
  const int R2 = Bb * S2 * K2;   // 262144
  const int R3 = Bb * K3;        // 4096
  const int T = 256;

  // d_in order = setup_inputs() dict insertion order, recursively:
  // 0: xyz; then sa1/sa2/sa3 layers (conv.w, conv.b, bn.g, bn.beta) x3 each;
  // then fc1.w, fc1.b, bn1.g, bn1.beta, fc2.w, fc2.b, bn2.g, bn2.beta, fc3.w, fc3.b.
  const float* xyz = (const float*)d_in[0];
  const float *W[9], *Bi[9], *Ga[9], *Be[9];
  for (int l = 0; l < 9; ++l) {
    W[l]  = (const float*)d_in[1 + l * 4 + 0];
    Bi[l] = (const float*)d_in[1 + l * 4 + 1];
    Ga[l] = (const float*)d_in[1 + l * 4 + 2];
    Be[l] = (const float*)d_in[1 + l * 4 + 3];
  }
  const float* fc1w = (const float*)d_in[37]; const float* fc1b = (const float*)d_in[38];
  const float* bn1g = (const float*)d_in[39]; const float* bn1b = (const float*)d_in[40];
  const float* fc2w = (const float*)d_in[41]; const float* fc2b = (const float*)d_in[42];
  const float* bn2g = (const float*)d_in[43]; const float* bn2b = (const float*)d_in[44];
  const float* fc3w = (const float*)d_in[45]; const float* fc3b = (const float*)d_in[46];

  // Bump allocator over d_ws (256B aligned slices).
  char* base = (char*)d_ws;
  size_t off = 0;
  auto alloc = [&](size_t bytes) -> void* {
    off = (off + 255) & ~(size_t)255;
    void* p = base + off;
    off += bytes;
    return p;
  };

  float* pts   = (float*)alloc((size_t)Bb * N * 3 * 4);
  int*   fidx1 = (int*)  alloc((size_t)Bb * S1 * 4);
  float* l1xyz = (float*)alloc((size_t)Bb * S1 * 3 * 4);
  int*   idx1  = (int*)  alloc((size_t)R1 * 4);
  int*   fidx2 = (int*)  alloc((size_t)Bb * S2 * 4);
  float* l2xyz = (float*)alloc((size_t)Bb * S2 * 3 * 4);
  int*   idx2  = (int*)  alloc((size_t)R2 * 4);
  float* l1pts = (float*)alloc((size_t)Bb * S1 * 128 * 4);
  float* l2pts = (float*)alloc((size_t)Bb * S2 * 256 * 4);
  float* meanb = (float*)alloc(1024 * 4);
  float* varb  = (float*)alloc(1024 * 4);
  const int   whN[6]   = {64 * 64, 128 * 64, 128 * 128, 256 * 128, 512 * 256, 1024 * 512};
  const float* wsrc[6] = {W[1], W[2], W[4], W[5], W[7], W[8]};
  half_t* wh[6];
  for (int i = 0; i < 6; ++i) wh[i] = (half_t*)alloc((size_t)whN[i] * 2);
  half_t* h1 = (half_t*)alloc((size_t)32 * 512 * 2);
  half_t* h2 = (half_t*)alloc((size_t)32 * 256 * 2);
  float* logits = (float*)alloc((size_t)32 * 40 * 4);
  // Ping-pong f16 activation buffers. bufB also hosts grouped inputs.
  half_t* bufA = (half_t*)alloc((size_t)R1 * 128 * 2);  // >= R2*256, R3*1024
  half_t* bufB = (half_t*)alloc((size_t)R2 * 131 * 2);  // >= R1*64, R2*128, R3*512

  auto cdiv = [](long a, long b) { return (int)((a + b - 1) / b); };

  // f16 weight copies for WMMA layers
  for (int i = 0; i < 6; ++i)
    cvt_f32_f16_kernel<<<cdiv(whN[i], T), T, 0, stream>>>(wsrc[i], wh[i], whN[i]);

  // ---------------- SA1 ----------------
  transpose_kernel<<<cdiv((long)Bb * N, T), T, 0, stream>>>(xyz, pts, Bb, N);
  fps_kernel<2048><<<Bb, 512, 0, stream>>>(pts, fidx1, S1);
  gather_xyz_kernel<<<cdiv((long)Bb * S1, T), T, 0, stream>>>(pts, fidx1, l1xyz, N, S1, Bb * S1);
  ballquery_kernel<<<cdiv((long)Bb * S1, T), T, 0, stream>>>(pts, l1xyz, idx1, Bb, N, S1, K1,
                                                             0.2f * 0.2f);
  group1_kernel<<<cdiv((long)R1, T), T, 0, stream>>>(pts, l1xyz, idx1, bufB, N, S1, K1, R1);
  lin_kernel<half_t, half_t><<<cdiv((long)R1 * 64, T), T, 0, stream>>>(bufB, W[0], Bi[0], bufA,
                                                                       R1, 3, 64);
  stats_kernel<<<64, T, 0, stream>>>(bufA, meanb, varb, R1, 64);
  bnrelu_kernel<<<cdiv((long)R1 * 64, T), T, 0, stream>>>(bufA, meanb, varb, Ga[0], Be[0],
                                                          (long)R1 * 64, 64);
  gemm_wmma_kernel<<<cdiv((long)(R1 / 16) * (64 / 64), 8), T, 0, stream>>>(bufA, wh[0], Bi[1],
                                                                           bufB, R1, 64, 64);
  stats_kernel<<<64, T, 0, stream>>>(bufB, meanb, varb, R1, 64);
  bnrelu_kernel<<<cdiv((long)R1 * 64, T), T, 0, stream>>>(bufB, meanb, varb, Ga[1], Be[1],
                                                          (long)R1 * 64, 64);
  gemm_wmma_kernel<<<cdiv((long)(R1 / 16) * (128 / 64), 8), T, 0, stream>>>(bufB, wh[1], Bi[2],
                                                                            bufA, R1, 64, 128);
  stats_kernel<<<128, T, 0, stream>>>(bufA, meanb, varb, R1, 128);
  bnrelu_kernel<<<cdiv((long)R1 * 128, T), T, 0, stream>>>(bufA, meanb, varb, Ga[2], Be[2],
                                                           (long)R1 * 128, 128);
  pool_kernel<<<cdiv((long)Bb * S1 * 128, T), T, 0, stream>>>(bufA, l1pts, Bb * S1, K1, 128);

  // ---------------- SA2 ----------------
  fps_kernel<512><<<Bb, 512, 0, stream>>>(l1xyz, fidx2, S2);
  gather_xyz_kernel<<<cdiv((long)Bb * S2, T), T, 0, stream>>>(l1xyz, fidx2, l2xyz, S1, S2,
                                                              Bb * S2);
  ballquery_kernel<<<cdiv((long)Bb * S2, T), T, 0, stream>>>(l1xyz, l2xyz, idx2, Bb, S1, S2, K2,
                                                             0.4f * 0.4f);
  group2_kernel<<<cdiv((long)R2, T), T, 0, stream>>>(l1xyz, l2xyz, l1pts, idx2, bufB, S1, S2, K2,
                                                     128, R2);
  lin_kernel<half_t, half_t><<<cdiv((long)R2 * 128, T), T, 0, stream>>>(bufB, W[3], Bi[3], bufA,
                                                                        R2, 131, 128);
  stats_kernel<<<128, T, 0, stream>>>(bufA, meanb, varb, R2, 128);
  bnrelu_kernel<<<cdiv((long)R2 * 128, T), T, 0, stream>>>(bufA, meanb, varb, Ga[3], Be[3],
                                                           (long)R2 * 128, 128);
  gemm_wmma_kernel<<<cdiv((long)(R2 / 16) * (128 / 64), 8), T, 0, stream>>>(bufA, wh[2], Bi[4],
                                                                            bufB, R2, 128, 128);
  stats_kernel<<<128, T, 0, stream>>>(bufB, meanb, varb, R2, 128);
  bnrelu_kernel<<<cdiv((long)R2 * 128, T), T, 0, stream>>>(bufB, meanb, varb, Ga[4], Be[4],
                                                           (long)R2 * 128, 128);
  gemm_wmma_kernel<<<cdiv((long)(R2 / 16) * (256 / 64), 8), T, 0, stream>>>(bufB, wh[3], Bi[5],
                                                                            bufA, R2, 128, 256);
  stats_kernel<<<256, T, 0, stream>>>(bufA, meanb, varb, R2, 256);
  bnrelu_kernel<<<cdiv((long)R2 * 256, T), T, 0, stream>>>(bufA, meanb, varb, Ga[5], Be[5],
                                                           (long)R2 * 256, 256);
  pool_kernel<<<cdiv((long)Bb * S2 * 256, T), T, 0, stream>>>(bufA, l2pts, Bb * S2, K2, 256);

  // ---------------- SA3 (group_all) ----------------
  group3_kernel<<<cdiv((long)R3 * 259, T), T, 0, stream>>>(l2xyz, l2pts, bufB, R3, 256);
  lin_kernel<half_t, half_t><<<cdiv((long)R3 * 256, T), T, 0, stream>>>(bufB, W[6], Bi[6], bufA,
                                                                        R3, 259, 256);
  stats_kernel<<<256, T, 0, stream>>>(bufA, meanb, varb, R3, 256);
  bnrelu_kernel<<<cdiv((long)R3 * 256, T), T, 0, stream>>>(bufA, meanb, varb, Ga[6], Be[6],
                                                           (long)R3 * 256, 256);
  gemm_wmma_kernel<<<cdiv((long)(R3 / 16) * (512 / 64), 8), T, 0, stream>>>(bufA, wh[4], Bi[7],
                                                                            bufB, R3, 256, 512);
  stats_kernel<<<512, T, 0, stream>>>(bufB, meanb, varb, R3, 512);
  bnrelu_kernel<<<cdiv((long)R3 * 512, T), T, 0, stream>>>(bufB, meanb, varb, Ga[7], Be[7],
                                                           (long)R3 * 512, 512);
  gemm_wmma_kernel<<<cdiv((long)(R3 / 16) * (1024 / 64), 8), T, 0, stream>>>(bufB, wh[5], Bi[8],
                                                                             bufA, R3, 512, 1024);
  stats_kernel<<<1024, T, 0, stream>>>(bufA, meanb, varb, R3, 1024);
  bnrelu_kernel<<<cdiv((long)R3 * 1024, T), T, 0, stream>>>(bufA, meanb, varb, Ga[8], Be[8],
                                                            (long)R3 * 1024, 1024);

  float* out = (float*)d_out;
  float* feat = out + 32 * 40;   // second tuple element: feat (32,1024)
  pool_kernel<<<cdiv((long)32 * 1024, T), T, 0, stream>>>(bufA, feat, 32, K3, 1024);

  // ---------------- Head ----------------
  lin_kernel<float, half_t><<<cdiv((long)32 * 512, T), T, 0, stream>>>(feat, fc1w, fc1b, h1, 32,
                                                                       1024, 512);
  stats_kernel<<<512, T, 0, stream>>>(h1, meanb, varb, 32, 512);
  bnrelu_kernel<<<cdiv((long)32 * 512, T), T, 0, stream>>>(h1, meanb, varb, bn1g, bn1b,
                                                           (long)32 * 512, 512);
  lin_kernel<half_t, half_t><<<cdiv((long)32 * 256, T), T, 0, stream>>>(h1, fc2w, fc2b, h2, 32,
                                                                        512, 256);
  stats_kernel<<<256, T, 0, stream>>>(h2, meanb, varb, 32, 256);
  bnrelu_kernel<<<cdiv((long)32 * 256, T), T, 0, stream>>>(h2, meanb, varb, bn2g, bn2b,
                                                           (long)32 * 256, 256);
  lin_kernel<half_t, float><<<cdiv((long)32 * 40, T), T, 0, stream>>>(h2, fc3w, fc3b, logits, 32,
                                                                      256, 40);
  logsoftmax_kernel<<<1, 32, 0, stream>>>(logits, out, 32, 40);
}
